// ConCareDEMinus_54889682043466
// MI455X (gfx1250) — compile-verified, matrix operand order checked by reading, not run on hardware
//
#include <hip/hip_runtime.h>

// ---------------------------------------------------------------------------
// ConCare (DE-minus) forward for MI455X / gfx1250, wave32 + WMMA bf16 + TDM.
// Sizes: B=128, T=128, F=64, H=128, D=128, HEADS=4 (dk=32), DFF=512, AH=8.
// Workspace ~294 MB (assumes ws_size >= ~310MB). d_out: 129 floats =
// sigmoid(logits)[128] ++ decov[1].
// ---------------------------------------------------------------------------

typedef __bf16 bf16_t;
typedef __attribute__((ext_vector_type(8)))  __bf16 v8bf;
typedef __attribute__((ext_vector_type(16))) __bf16 v16bf;
typedef __attribute__((ext_vector_type(8)))  float  v8f;
typedef __attribute__((ext_vector_type(4))) unsigned int u32x4;
typedef __attribute__((ext_vector_type(4))) int i32x4;
typedef __attribute__((ext_vector_type(8))) int i32x8;

#define B_  128
#define T_  128
#define F_  64
#define H_  128
#define G3_ 384
#define D_  128
#define DFF_ 512

// ---- fast transcendentals (hardware v_tanh_f32 / v_exp_f32 / v_rcp_f32) ----
#if __has_builtin(__builtin_amdgcn_tanhf)
__device__ __forceinline__ float fast_tanh(float v) { return __builtin_amdgcn_tanhf(v); }
#else
__device__ __forceinline__ float fast_tanh(float v) { return tanhf(v); }
#endif
#if __has_builtin(__builtin_amdgcn_rcpf)
__device__ __forceinline__ float fast_rcp(float v) { return __builtin_amdgcn_rcpf(v); }
#else
__device__ __forceinline__ float fast_rcp(float v) { return 1.f / v; }
#endif
#if __has_builtin(__builtin_amdgcn_exp2f)
__device__ __forceinline__ float fast_exp(float v) { return __builtin_amdgcn_exp2f(v * 1.44269504088896340736f); }
#else
__device__ __forceinline__ float fast_exp(float v) { return __expf(v); }
#endif
__device__ __forceinline__ float sigm(float v) { return fast_rcp(1.f + fast_exp(-v)); }

// A-fragment (16x32 bf16) for v_wmma_f32_16x16x32_bf16 from bf16 matrix, row
// stride `ld` (elements). Per ISA 7.12.2:
//  lanes 0-15 : row=rbase+lane, elems 0..7 -> K=k0+0..7,  8..15 -> K=k0+16..23
//  lanes 16-31: row=rbase+lane-16, elems 0..7 -> K=k0+8..15, 8..15 -> K=k0+24..31
// Requires ld % 8 == 0 and 16B-aligned base (chunks become b128 loads).
__device__ __forceinline__ v16bf frag16(const bf16_t* base, int ld, int rbase, int k0, int lane) {
    int hf = lane >> 4;
    int l  = lane & 15;
    const bf16_t* p = base + (size_t)(rbase + l) * ld + k0 + hf * 8;
    union { v16bf v; v8bf h[2]; } u;
    u.h[0] = *(const v8bf*)p;
    u.h[1] = *(const v8bf*)(p + 16);
    return u.v;
}

// ---------------------------------------------------------------------------
// Kernel A: batched GRU scan. 1 block per feature f, 256 threads = 8 waves.
// h state in LDS; W_hh[f] B-fragments register-resident for the whole scan.
// x column double-buffered in LDS per step.
// ---------------------------------------------------------------------------
__global__ __launch_bounds__(256) void gru_scan_kernel(
    const float* __restrict__ x,      // [B,T,F]
    const float* __restrict__ W_ih,   // [F,3H]
    const float* __restrict__ b_ih,   // [F,3H]
    const float* __restrict__ W_hh,   // [F,3H,H]
    const float* __restrict__ b_hh,   // [F,3H]
    bf16_t* __restrict__ hs_out)      // [F,T,B,H] bf16
{
    const int f    = blockIdx.x;
    const int tid  = threadIdx.x;
    const int wave = tid >> 5;
    const int lane = tid & 31;
    const int l    = lane & 15;
    const int hf   = lane >> 4;
    const int col  = wave * 16 + l;   // this lane's H-column
    constexpr int HP = H_ + 8;

    __shared__ bf16_t hsh[B_ * HP];                 // ~34.8 KB
    __shared__ float  sWih[G3_], sbih[G3_], sbhh[G3_];
    __shared__ float  sx[2][B_];                    // per-step x column, 2-deep

    for (int i = tid; i < G3_; i += 256) {
        sWih[i] = W_ih[f * G3_ + i];
        sbih[i] = b_ih[f * G3_ + i];
        sbhh[i] = b_hh[f * G3_ + i];
    }
    for (int i = tid; i < B_ * HP; i += 256) hsh[i] = (bf16_t)0.f;
    __syncthreads();

    const float wir = sWih[col], wiz = sWih[H_ + col], win = sWih[2 * H_ + col];
    const float bir = sbih[col], biz = sbih[H_ + col], bin_ = sbih[2 * H_ + col];
    const float bhr = sbhh[col], bhz = sbhh[H_ + col], bhn = sbhh[2 * H_ + col];

    // W_hh[f] B-fragments: B[k][n=g] = W_hh[f][g][k] (contiguous in k).
    v16bf Bfrag[3][4];
    const float* Wf = W_hh + (size_t)f * G3_ * H_;
#pragma unroll
    for (int g = 0; g < 3; ++g) {
#pragma unroll
        for (int kb = 0; kb < 4; ++kb) {
            const float* p = Wf + (size_t)(g * H_ + col) * H_ + kb * 32 + hf * 8;
            v16bf v;
#pragma unroll
            for (int e = 0; e < 8; ++e) v[e] = (bf16_t)p[e];
#pragma unroll
            for (int e = 0; e < 8; ++e) v[e + 8] = (bf16_t)p[e + 16];
            Bfrag[g][kb] = v;
        }
    }

    for (int t = 0; t < T_; ++t) {
        // Stage x[:, t, f]; parity buffer keeps laggard waves (still in step
        // t-1 gate math) reading the other slot. Visibility ordered by the
        // bt=0 barrier below.
        if (tid < B_) sx[t & 1][tid] = x[(size_t)tid * (T_ * F_) + t * F_ + f];
        const float* sxc = sx[t & 1];

        for (int bt = 0; bt < 8; ++bt) {
            v8f ar = {}, az = {}, an = {};
#pragma unroll
            for (int kb = 0; kb < 4; ++kb) {
                v16bf a = frag16(hsh, HP, bt * 16, kb * 32, lane);
                ar = __builtin_amdgcn_wmma_f32_16x16x32_bf16(false, a, false, Bfrag[0][kb], (short)0, ar, false, false);
                az = __builtin_amdgcn_wmma_f32_16x16x32_bf16(false, a, false, Bfrag[1][kb], (short)0, az, false, false);
                an = __builtin_amdgcn_wmma_f32_16x16x32_bf16(false, a, false, Bfrag[2][kb], (short)0, an, false, false);
            }
            // All reads of rows [bt*16, bt*16+16) complete before writes.
            __syncthreads();
#pragma unroll
            for (int j = 0; j < 8; ++j) {
                int row    = bt * 16 + hf * 8 + j;            // batch index
                float xv   = sxc[row];
                float r    = sigm(xv * wir + bir + ar[j] + bhr);
                float z    = sigm(xv * wiz + biz + az[j] + bhz);
                float n    = fast_tanh(xv * win + bin_ + r * (an[j] + bhn));
                float hold = (float)hsh[row * HP + col];
                float hnew = (1.f - z) * n + z * hold;
                hsh[row * HP + col] = (bf16_t)hnew;
                hs_out[(((size_t)f * T_ + t) * B_ + row) * H_ + col] = (bf16_t)hnew;
            }
        }
    }
}

// ---------------------------------------------------------------------------
// Kernel B: per-feature additive attention over time. 1 block per (b,f).
// hs tile (128x128 bf16, contiguous-h rows with stride B*H) staged into LDS
// with the Tensor Data Mover where available.
// ---------------------------------------------------------------------------
#if defined(__gfx1250__) && __has_builtin(__builtin_amdgcn_tensor_load_to_lds) && __has_builtin(__builtin_amdgcn_s_wait_tensorcnt)
#define HAVE_TDM 1
#else
#define HAVE_TDM 0
#endif

__global__ __launch_bounds__(128) void feat_attn_kernel(
    const bf16_t* __restrict__ hs,    // [F,T,B,H]
    const float* __restrict__ Wq_f,   // [F,H,8]
    const float* __restrict__ bq_f,   // [F,8]
    const float* __restrict__ Wk_f,   // [F,H,8]
    const float* __restrict__ bk_f,   // [F,8]
    const float* __restrict__ v_f,    // [F,8]
    float* __restrict__ feats,        // [B*F,H] f32
    bf16_t* __restrict__ featsb)      // [B*F,H] bf16
{
    const int b = blockIdx.x >> 6;
    const int f = blockIdx.x & 63;
    const int tid = threadIdx.x;

    __shared__ bf16_t sh[T_ * H_];    // 32 KB staged hs tile, [t][h]
    __shared__ float sWq[H_ * 8], sWk[H_ * 8];
    __shared__ float sq[8], sv[8], se[T_], sa[T_], red[T_];

    const bf16_t* gtile = hs + (size_t)f * T_ * B_ * H_ + (size_t)b * H_;  // (t,h) @ + t*B*H + h

#if HAVE_TDM
    if (tid < 32) {   // wave 0 issues the tensor DMA (EXEC ignored; 1 issue/wave)
        unsigned long long ga = (unsigned long long)(uintptr_t)gtile;
        unsigned int lds = (unsigned int)(uintptr_t)&sh[0];   // low 32 bits = LDS byte addr
        u32x4 g0 = {0u, 0u, 0u, 0u};
        g0[0] = 1u;                                            // count=1, user mode
        g0[1] = lds;                                           // lds_addr
        g0[2] = (unsigned int)(ga & 0xffffffffu);              // global_addr[31:0]
        g0[3] = (unsigned int)((ga >> 32) & 0x01ffffffu) | (2u << 30);  // [56:32] + type=2
        i32x8 g1 = {0, 0, 0, 0, 0, 0, 0, 0};
        g1[0] = (int)(1u << 16);                  // data_size=1 (2B); mask/pad/iter=0
        g1[1] = (int)((unsigned)H_ << 16);        // tensor_dim0[15:0] @ bits 63:48
        g1[2] = (int)((unsigned)T_ << 16);        // tensor_dim1[15:0] @ bits 95:80
        g1[3] = (int)((unsigned)H_ << 16);        // tile_dim0 @ bits 127:112
        g1[4] = (int)(unsigned)T_;                // tile_dim1 @ bits 143:128 (tile_dim2=0)
        g1[5] = (int)(unsigned)(B_ * H_);         // tensor_dim0_stride[31:0]
        i32x4 gz = {0, 0, 0, 0};
#if __clang_major__ >= 23
        i32x8 gz8 = {0, 0, 0, 0, 0, 0, 0, 0};
        __builtin_amdgcn_tensor_load_to_lds(g0, g1, gz, gz, gz8, 0);
#else
        __builtin_amdgcn_tensor_load_to_lds(g0, g1, gz, gz, 0);
#endif
        __builtin_amdgcn_s_wait_tensorcnt(0);
    }
#else
    for (int i = tid; i < T_ * H_; i += 128)
        sh[i] = gtile[(size_t)(i >> 7) * (B_ * H_) + (i & 127)];
#endif

    for (int i = tid; i < H_ * 8; i += 128) { sWq[i] = Wq_f[f * H_ * 8 + i]; sWk[i] = Wk_f[f * H_ * 8 + i]; }
    if (tid < 8) sv[tid] = v_f[f * 8 + tid];
    __syncthreads();

    if (tid < 8) {
        float acc = bq_f[f * 8 + tid];
        const bf16_t* hrow = &sh[(T_ - 1) * H_];
        for (int h = 0; h < H_; ++h) acc += (float)hrow[h] * sWq[h * 8 + tid];
        sq[tid] = acc;
    }
    __syncthreads();

    {   // e_t for this thread's timestep
        const bf16_t* hrow = &sh[tid * H_];
        float k[8];
#pragma unroll
        for (int a = 0; a < 8; ++a) k[a] = bk_f[f * 8 + a];
        for (int h = 0; h < H_; ++h) {
            float hv = (float)hrow[h];
#pragma unroll
            for (int a = 0; a < 8; ++a) k[a] += hv * sWk[h * 8 + a];
        }
        float e = 0.f;
#pragma unroll
        for (int a = 0; a < 8; ++a) e += sv[a] * fast_tanh(sq[a] + k[a]);
        se[tid] = e;
    }
    __syncthreads();

    // softmax over T=128
    red[tid] = se[tid]; __syncthreads();
    for (int s = 64; s > 0; s >>= 1) { if (tid < s) red[tid] = fmaxf(red[tid], red[tid + s]); __syncthreads(); }
    float m = red[0]; __syncthreads();
    float ex = fast_exp(se[tid] - m);
    red[tid] = ex; __syncthreads();
    for (int s = 64; s > 0; s >>= 1) { if (tid < s) red[tid] += red[tid + s]; __syncthreads(); }
    sa[tid] = ex * fast_rcp(red[0]);
    __syncthreads();

    {   // feats[b,f,h] = sum_t a_t * hs[t,h], thread = h
        float acc = 0.f;
        for (int t = 0; t < T_; ++t) acc += sa[t] * (float)sh[t * H_ + tid];
        size_t o = ((size_t)b * F_ + f) * H_ + tid;
        feats[o]  = acc;
        featsb[o] = (bf16_t)acc;
    }
}

// ---------------------------------------------------------------------------
// Generic bf16-WMMA GEMM: C[M,N] = act(A[M,K]@B[K,N] + bias + residual)
// A bf16 (global), B f32 weights. Both staged into LDS per 128-K chunk:
// A as [m][k] b128 copies, B transposed to [n][k] (coalesced f32 reads).
// grid = (M/64, N/128), block 256 = 8 waves; wave w -> 16 cols, 4 row tiles.
// ---------------------------------------------------------------------------
__global__ __launch_bounds__(256) void gemm_wmma_kernel(
    const bf16_t* __restrict__ A, const float* __restrict__ Bw,
    const float* __restrict__ bias, const float* __restrict__ residual,
    float* __restrict__ Cf, bf16_t* __restrict__ Cb,
    int M, int N, int K, int relu)
{
    constexpr int SP = 136;   // padded LDS row stride (elements)
    __shared__ bf16_t sA[64 * SP];     // ~17.4 KB
    __shared__ bf16_t sBT[128 * SP];   // ~34.8 KB, [n][k]

    const int Mbase = blockIdx.x * 64;
    const int Nbase = blockIdx.y * 128;
    const int tid   = threadIdx.x;
    const int wave  = tid >> 5;
    const int lane  = tid & 31;
    const int l     = lane & 15;
    const int hf    = lane >> 4;
    const int col   = Nbase + wave * 16 + l;
    const float bv  = bias ? bias[col] : 0.f;

    v8f acc[4] = {{}, {}, {}, {}};

    for (int kc = 0; kc < K / 128; ++kc) {
        __syncthreads();   // previous chunk's fragment reads complete
        // Stage A chunk: 64 rows x 128 k, vectorized 8xbf16 copies.
        for (int i = tid; i < 64 * 16; i += 256) {
            int m = i >> 4, vk = i & 15;
            *(v8bf*)&sA[m * SP + vk * 8] =
                *(const v8bf*)(A + (size_t)(Mbase + m) * K + kc * 128 + vk * 8);
        }
        // Stage B chunk transposed: coalesced f32 reads, bf16 LDS scatter.
        for (int i = tid; i < 128 * 128; i += 256) {
            int k = i >> 7, n = i & 127;
            sBT[n * SP + k] = (bf16_t)Bw[(size_t)(kc * 128 + k) * N + Nbase + n];
        }
        __syncthreads();
#pragma unroll
        for (int mt = 0; mt < 4; ++mt) {
#pragma unroll
            for (int kb = 0; kb < 4; ++kb) {
                v16bf a  = frag16(sA,  SP, mt * 16,   kb * 32, lane);
                v16bf bb = frag16(sBT, SP, wave * 16, kb * 32, lane);
                acc[mt] = __builtin_amdgcn_wmma_f32_16x16x32_bf16(false, a, false, bb, (short)0, acc[mt], false, false);
            }
        }
    }
#pragma unroll
    for (int mt = 0; mt < 4; ++mt) {
#pragma unroll
        for (int j = 0; j < 8; ++j) {
            int row = Mbase + mt * 16 + hf * 8 + j;
            float v = acc[mt][j] + bv;
            if (residual) v += residual[(size_t)row * N + col];
            if (relu) v = fmaxf(v, 0.f);
            size_t o = (size_t)row * N + col;
            if (Cf) Cf[o] = v;
            if (Cb) Cb[o] = (bf16_t)v;
        }
    }
}

// ---------------------------------------------------------------------------
// MHSA over F=64 channels, per (b, head). 64 threads = query rows. dk=32.
// ---------------------------------------------------------------------------
__global__ __launch_bounds__(64) void mhsa_kernel(
    const float* __restrict__ Q, const float* __restrict__ Km, const float* __restrict__ Vm,
    bf16_t* __restrict__ ctxb)   // [B*F, D] bf16
{
    const int b = blockIdx.x >> 2;
    const int h = blockIdx.x & 3;
    const int i = threadIdx.x;
    const float scale = 0.1767766952966369f;  // 1/sqrt(32)

    float q[32];
    const float* qr = Q + ((size_t)b * F_ + i) * D_ + h * 32;
#pragma unroll
    for (int d = 0; d < 32; ++d) q[d] = qr[d];

    float s[64];
    float m = -1e30f;
    for (int j = 0; j < 64; ++j) {
        const float* kr = Km + ((size_t)b * F_ + j) * D_ + h * 32;
        float acc = 0.f;
#pragma unroll
        for (int d = 0; d < 32; ++d) acc += q[d] * kr[d];
        acc *= scale;
        s[j] = acc;
        m = fmaxf(m, acc);
    }
    float Z = 0.f;
    for (int j = 0; j < 64; ++j) { s[j] = fast_exp(s[j] - m); Z += s[j]; }
    float inv = fast_rcp(Z);

    float c[32];
#pragma unroll
    for (int d = 0; d < 32; ++d) c[d] = 0.f;
    for (int j = 0; j < 64; ++j) {
        const float* vr = Vm + ((size_t)b * F_ + j) * D_ + h * 32;
        float aj = s[j] * inv;
#pragma unroll
        for (int d = 0; d < 32; ++d) c[d] += aj * vr[d];
    }
    bf16_t* o = ctxb + ((size_t)b * F_ + i) * D_ + h * 32;
#pragma unroll
    for (int d = 0; d < 32; ++d) o[d] = (bf16_t)c[d];
}

// ---------------------------------------------------------------------------
// DeCov: per f, cov = mc@mc^T/(D-1); accumulate 0.5 * sum_{b!=c} cov^2.
// Gram matrix via WMMA (B-fragment of mc^T == A-fragment pattern of mc).
// ---------------------------------------------------------------------------
__global__ __launch_bounds__(256) void decov_kernel(
    const bf16_t* __restrict__ ctxb,  // [B*F, D]
    float* __restrict__ out_decov)
{
    const int f = blockIdx.x;
    constexpr int HP = D_ + 8;
    __shared__ bf16_t mc[B_ * HP];
    __shared__ float rbuf[256];
    const int tid = threadIdx.x;

    if (tid < B_) {
        const bf16_t* src = ctxb + ((size_t)tid * F_ + f) * D_;
        float mean = 0.f;
        for (int d = 0; d < D_; ++d) mean += (float)src[d];
        mean *= (1.f / (float)D_);
        for (int d = 0; d < D_; ++d) mc[tid * HP + d] = (bf16_t)((float)src[d] - mean);
    }
    __syncthreads();

    const int wave = tid >> 5, lane = tid & 31, l = lane & 15, hf = lane >> 4;
    float ssum = 0.f;
    for (int nt = 0; nt < 8; ++nt) {
        v8f acc = {};
#pragma unroll
        for (int kb = 0; kb < 4; ++kb) {
            v16bf a  = frag16(mc, HP, wave * 16, kb * 32, lane);
            v16bf bb = frag16(mc, HP, nt * 16,  kb * 32, lane);
            acc = __builtin_amdgcn_wmma_f32_16x16x32_bf16(false, a, false, bb, (short)0, acc, false, false);
        }
        int colg = nt * 16 + l;
#pragma unroll
        for (int j = 0; j < 8; ++j) {
            int rowg = wave * 16 + hf * 8 + j;
            float v = acc[j] * (1.f / (float)(D_ - 1));
            if (rowg != colg) ssum += v * v;
        }
    }
    rbuf[tid] = ssum; __syncthreads();
    for (int s = 128; s > 0; s >>= 1) { if (tid < s) rbuf[tid] += rbuf[tid + s]; __syncthreads(); }
    if (tid == 0) atomicAdd(out_decov, 0.5f * rbuf[0]);
}

// ---------------------------------------------------------------------------
// Final attention head + MLP + sigmoid. 1 block per b, 128 threads.
// ---------------------------------------------------------------------------
__global__ __launch_bounds__(128) void final_kernel(
    const float* __restrict__ C2,     // [B*F, D]
    const float* __restrict__ Fq, const float* __restrict__ fbq,
    const float* __restrict__ Fk, const float* __restrict__ fbk,
    const float* __restrict__ Fv, const float* __restrict__ fbv,
    const float* __restrict__ O0, const float* __restrict__ ob0,
    const float* __restrict__ O1, const float* __restrict__ ob1,
    float* __restrict__ out)          // [B] sigmoid logits
{
    const int b = blockIdx.x;
    const int tid = threadIdx.x;
    __shared__ float sC[F_ * D_];
    __shared__ float sfq[H_], sfa[F_], spool[H_], shid[H_];

    for (int i = tid; i < F_ * D_; i += 128) sC[i] = C2[(size_t)b * F_ * D_ + i];
    __syncthreads();

    {   // fq[h] from last feature channel
        float acc = fbq[tid];
        for (int d = 0; d < D_; ++d) acc += sC[(F_ - 1) * D_ + d] * Fq[d * H_ + tid];
        sfq[tid] = acc;
    }
    __syncthreads();

    if (tid < F_) {
        float e = 0.f;
        for (int h = 0; h < H_; ++h) {
            float kfh = fbk[h];
            for (int d = 0; d < D_; ++d) kfh += sC[tid * D_ + d] * Fk[d * H_ + h];
            e += sfq[h] * kfh;
        }
        sfa[tid] = e;
    }
    __syncthreads();
    if (tid == 0) {
        float m = -1e30f;
        for (int j = 0; j < F_; ++j) m = fmaxf(m, sfa[j]);
        float Z = 0.f;
        for (int j = 0; j < F_; ++j) { sfa[j] = fast_exp(sfa[j] - m); Z += sfa[j]; }
        float inv = fast_rcp(Z);
        for (int j = 0; j < F_; ++j) sfa[j] *= inv;
    }
    __syncthreads();

    {   // pooled[h]
        float acc = 0.f;
        for (int j = 0; j < F_; ++j) {
            float fvv = fbv[tid];
            for (int d = 0; d < D_; ++d) fvv += sC[j * D_ + d] * Fv[d * H_ + tid];
            acc += sfa[j] * fvv;
        }
        spool[tid] = acc;
    }
    __syncthreads();
    {
        float acc = ob0[tid];
        for (int d = 0; d < H_; ++d) acc += spool[d] * O0[d * H_ + tid];
        shid[tid] = fmaxf(acc, 0.f);
    }
    __syncthreads();
    if (tid == 0) {
        float acc = ob1[0];
        for (int h = 0; h < H_; ++h) acc += shid[h] * O1[h];
        out[b] = sigm(acc);
    }
}

// ---------------------------------------------------------------------------
extern "C" void kernel_launch(void* const* d_in, const int* in_sizes, int n_in,
                              void* d_out, int out_size, void* d_ws, size_t ws_size,
                              hipStream_t stream) {
    (void)in_sizes; (void)n_in; (void)out_size; (void)ws_size;
    const float* x    = (const float*)d_in[0];
    const float* W_ih = (const float*)d_in[2];
    const float* b_ih = (const float*)d_in[3];
    const float* W_hh = (const float*)d_in[4];
    const float* b_hh = (const float*)d_in[5];
    const float* Wq_f = (const float*)d_in[6];
    const float* bq_f = (const float*)d_in[7];
    const float* Wk_f = (const float*)d_in[8];
    const float* bk_f = (const float*)d_in[9];
    const float* v_f  = (const float*)d_in[10];
    const float* Wq = (const float*)d_in[11];  const float* bq = (const float*)d_in[12];
    const float* Wk = (const float*)d_in[13];  const float* bk = (const float*)d_in[14];
    const float* Wv = (const float*)d_in[15];  const float* bv = (const float*)d_in[16];
    const float* Wo = (const float*)d_in[17];  const float* bo = (const float*)d_in[18];
    const float* W1 = (const float*)d_in[19];  const float* b1 = (const float*)d_in[20];
    const float* W2 = (const float*)d_in[21];  const float* b2 = (const float*)d_in[22];
    const float* Fq = (const float*)d_in[23];  const float* fbq = (const float*)d_in[24];
    const float* Fk = (const float*)d_in[25];  const float* fbk = (const float*)d_in[26];
    const float* Fv = (const float*)d_in[27];  const float* fbv = (const float*)d_in[28];
    const float* O0 = (const float*)d_in[29];  const float* ob0 = (const float*)d_in[30];
    const float* O1 = (const float*)d_in[31];  const float* ob1 = (const float*)d_in[32];
    float* out = (float*)d_out;

    char* ws = (char*)d_ws;
    bf16_t* hs     = (bf16_t*)(ws + 0);            // 256 MB
    float*  feats  = (float*) (ws + 268435456);    //   4 MB
    bf16_t* featsb = (bf16_t*)(ws + 272629760);    //   2 MB
    float*  Qf     = (float*) (ws + 274726912);    //   4 MB
    float*  Kf     = (float*) (ws + 278921216);    //   4 MB
    float*  Vf     = (float*) (ws + 283115520);    //   4 MB
    bf16_t* ctxb   = (bf16_t*)(ws + 287309824);    //   2 MB
    float*  c1f    = (float*) (ws + 289406976);    //   4 MB (contexts after Wo+residual)
    bf16_t* c1b    = (bf16_t*)(ws + 293601280);    //   2 MB
    bf16_t* hdnb   = (bf16_t*)(ws + 295698432);    //   8 MB (relu hidden, bf16)
    float*  c2f    = (float*) (ws + 304087040);    //   4 MB (contexts after FFN)

    hipMemsetAsync(out + 128, 0, sizeof(float), stream);  // decov accumulator

    gru_scan_kernel<<<F_, 256, 0, stream>>>(x, W_ih, b_ih, W_hh, b_hh, hs);
    feat_attn_kernel<<<B_ * F_, 128, 0, stream>>>(hs, Wq_f, bq_f, Wk_f, bk_f, v_f, feats, featsb);

    gemm_wmma_kernel<<<dim3(128, 1), 256, 0, stream>>>(featsb, Wq, bq, nullptr, Qf, nullptr, B_ * F_, D_, D_, 0);
    gemm_wmma_kernel<<<dim3(128, 1), 256, 0, stream>>>(featsb, Wk, bk, nullptr, Kf, nullptr, B_ * F_, D_, D_, 0);
    gemm_wmma_kernel<<<dim3(128, 1), 256, 0, stream>>>(featsb, Wv, bv, nullptr, Vf, nullptr, B_ * F_, D_, D_, 0);

    mhsa_kernel<<<B_ * 4, 64, 0, stream>>>(Qf, Kf, Vf, ctxb);
    decov_kernel<<<F_, 256, 0, stream>>>(ctxb, out + 128);

    // contexts = feats + ctx@Wo + bo
    gemm_wmma_kernel<<<dim3(128, 1), 256, 0, stream>>>(ctxb, Wo, bo, feats, c1f, c1b, B_ * F_, D_, D_, 0);
    // hdn = relu(contexts@W1 + b1)
    gemm_wmma_kernel<<<dim3(128, 4), 256, 0, stream>>>(c1b, W1, b1, nullptr, nullptr, hdnb, B_ * F_, DFF_, D_, 1);
    // contexts2 = contexts + hdn@W2 + b2
    gemm_wmma_kernel<<<dim3(128, 1), 256, 0, stream>>>(hdnb, W2, b2, c1f, c2f, nullptr, B_ * F_, D_, DFF_, 0);

    final_kernel<<<B_, 128, 0, stream>>>(c2f, Fq, fbq, Fk, fbk, Fv, fbv, O0, ob0, O1, ob1, out);
}
